// W4A16TVMMLP_30416958390340
// MI455X (gfx1250) — matrix operand order repaired and checked by reading, not compiled, not run on hardware
//
#include <hip/hip_runtime.h>
#include <math.h>

#define HID   2048
#define MLPD  16384
#define NTOK  128
#define LDSW  40    // padded LDS row stride in halfs (80B: 16B-aligned, bank-conflict-free)

typedef __attribute__((ext_vector_type(16))) _Float16 v16h;
typedef __attribute__((ext_vector_type(8)))  _Float16 v8h;
typedef __attribute__((ext_vector_type(4)))  _Float16 v4h;
typedef __attribute__((ext_vector_type(2)))  _Float16 v2h;
typedef __attribute__((ext_vector_type(8)))  float    v8f;
typedef __attribute__((ext_vector_type(4)))  float    v4f;

__device__ __constant__ float kNvfp4Lut[16] = {
    0.0f,  0.5f,  1.0f,  1.5f,  2.0f,  3.0f,  4.0f,  6.0f,
    0.0f, -0.5f, -1.0f, -1.5f, -2.0f, -3.0f, -4.0f, -6.0f };

// Byte -> packed f16 pair (low nibble in h[0]).
__device__ inline unsigned lut_entry(int byte) {
    union { v2h h; unsigned u; } pk;
    pk.h[0] = (_Float16)kNvfp4Lut[byte & 15];
    pk.h[1] = (_Float16)kNvfp4Lut[(byte >> 4) & 15];
    return pk.u;
}

__device__ inline v2h scale_mul(unsigned lutword, float scale) {
    union { unsigned u; v2h h; } e; e.u = lutword;
    const _Float16 s = (_Float16)scale;
    v2h sv; sv[0] = s; sv[1] = s;
    return e.h * sv;                       // v_pk_mul_f16
}

// ---------------------------------------------------------------------------
// Kernel 0: x fp32 -> f16 once.
// ---------------------------------------------------------------------------
__global__ __launch_bounds__(256) void cvt_x_kernel(
    const float* __restrict__ x, _Float16* __restrict__ xh)
{
    const int i = (blockIdx.x * 256 + threadIdx.x) * 4;
    const v4f f = *(const v4f*)(x + i);
    v4h h;
#pragma unroll
    for (int j = 0; j < 4; ++j) h[j] = (_Float16)f[j];
    *(v4h*)(xh + i) = h;
}

// ---------------------------------------------------------------------------
// Kernel 1: gate/up GEMMs + tanh-GELU -> f16 inter.
// grid = MLPD/64; 64-wide N tile (4 sub-tiles); 8 wave32, wave = 16 tokens.
// Double-buffered LDS, ONE barrier per K-step:
//   iter i: [issue i+1 global loads][WMMA from buf(i&1)][decode i+1 -> buf(~i&1)][barrier]
// ---------------------------------------------------------------------------
__global__ __launch_bounds__(256) void mlp_gateup_kernel(
    const _Float16* __restrict__ xh,
    const int*   __restrict__ gW, const float* __restrict__ gS,
    const int*   __restrict__ uW, const float* __restrict__ uS,
    _Float16*    __restrict__ inter)
{
    __shared__ _Float16 ldsG[2][64 * LDSW];
    __shared__ _Float16 ldsU[2][64 * LDSW];
    __shared__ unsigned lut256[256];

    const int tid = threadIdx.x;
    lut256[tid] = lut_entry(tid);

    const int n0   = blockIdx.x * 64;
    const int lane = tid & 31;
    const int t0   = (tid >> 5) * 16;

    const int rbase = tid >> 4;            // dequant rows rbase + {0,16,32,48}
    const int j     = tid & 15;

    const int m     = lane & 15;
    const int khalf = lane >> 4;
    const int token = t0 + m;

    const int bn   = lane & 15;
    const int koff = (lane >> 4) * 16;

    size_t rw4[4], rs4[4];
#pragma unroll
    for (int s = 0; s < 4; ++s) {
        const int row = rbase + 16 * s;
        rw4[s] = (size_t)(n0 + row) * (HID / 2);
        rs4[s] = (size_t)(n0 + row) * (HID / 32);
    }

    v8f accG[4] = {{}, {}, {}, {}};
    v8f accU[4] = {{}, {}, {}, {}};

    // ---- prologue: load + decode K-step 0 into buffer 0 ----
    int   pg[4], pu[4];
    float sg[4], su[4];
#pragma unroll
    for (int s = 0; s < 4; ++s) {
        pg[s] = gW[rw4[s] + j];
        pu[s] = uW[rw4[s] + j];
        sg[s] = gS[rs4[s]];
        su[s] = uS[rs4[s]];
    }
    __syncthreads();                       // lut256 ready
    {
        unsigned lwG[4], lwU[4];
#pragma unroll
        for (int s = 0; s < 4; ++s) {      // batch LUT loads, then muls
            lwG[s] = lut256[pg[s] & 255];
            lwU[s] = lut256[pu[s] & 255];
        }
#pragma unroll
        for (int s = 0; s < 4; ++s) {
            const int row = rbase + 16 * s;
            *(v2h*)&ldsG[0][row * LDSW + 2 * j] = scale_mul(lwG[s], sg[s]);
            *(v2h*)&ldsU[0][row * LDSW + 2 * j] = scale_mul(lwU[s], su[s]);
        }
    }
    __syncthreads();                       // buffer 0 visible

    const int iters = HID / 32;
    for (int it = 0; it < iters; ++it) {
        const int k0   = it * 32;
        const int cur  = it & 1;
        const bool more = (it + 1) < iters;

        // ---- 1. issue NEXT K-step's packed loads (hidden under WMMAs) ----
        if (more) {
#pragma unroll
            for (int s = 0; s < 4; ++s) {
                pg[s] = gW[rw4[s] + ((k0 + 32) >> 1) + j];
                pu[s] = uW[rw4[s] + ((k0 + 32) >> 1) + j];
                sg[s] = gS[rs4[s] + ((k0 + 32) >> 5)];
                su[s] = uS[rs4[s] + ((k0 + 32) >> 5)];
            }
        }

        // ---- 2. A fragment + 8 WMMAs from buf[cur] ----
        const _Float16* xp = xh + (size_t)token * HID + k0 + khalf * 16;
        const v8h a0 = *(const v8h*)xp;
        const v8h a1 = *(const v8h*)(xp + 8);
        v16h a;
#pragma unroll
        for (int i = 0; i < 8; ++i) { a[i] = a0[i]; a[8 + i] = a1[i]; }

#pragma unroll
        for (int nt = 0; nt < 4; ++nt) {
            const v16h bg = *(const v16h*)&ldsG[cur][(nt * 16 + bn) * LDSW + koff];
            accG[nt] = __builtin_amdgcn_wmma_f32_16x16x32_f16(
                false, a, false, bg, (short)0, accG[nt], false, false);
        }
#pragma unroll
        for (int nt = 0; nt < 4; ++nt) {
            const v16h bu = *(const v16h*)&ldsU[cur][(nt * 16 + bn) * LDSW + koff];
            accU[nt] = __builtin_amdgcn_wmma_f32_16x16x32_f16(
                false, a, false, bu, (short)0, accU[nt], false, false);
        }

        // ---- 3. decode next step into the other buffer (no conflict) ----
        if (more) {
            unsigned lwG[4], lwU[4];
#pragma unroll
            for (int s = 0; s < 4; ++s) {
                lwG[s] = lut256[pg[s] & 255];
                lwU[s] = lut256[pu[s] & 255];
            }
#pragma unroll
            for (int s = 0; s < 4; ++s) {
                const int row = rbase + 16 * s;
                *(v2h*)&ldsG[cur ^ 1][row * LDSW + 2 * j] = scale_mul(lwG[s], sg[s]);
                *(v2h*)&ldsU[cur ^ 1][row * LDSW + 2 * j] = scale_mul(lwU[s], su[s]);
            }
        }

        __syncthreads();                   // single barrier per K-step
    }

    // ---- epilogue: tanh-GELU(gate) * up -> f16 inter ----
    const int cn    = lane & 15;
    const int mbase = t0 + ((lane >> 4) << 3);
#pragma unroll
    for (int nt = 0; nt < 4; ++nt) {
#pragma unroll
        for (int rr = 0; rr < 8; ++rr) {
            const float g  = accG[nt][rr];
            const float u  = accU[nt][rr];
            const float g3 = g * g * g;
            const float gelu = 0.5f * g *
                (1.0f + tanhf(0.7978845608028654f * (g + 0.044715f * g3)));
            inter[(size_t)(mbase + rr) * MLPD + n0 + nt * 16 + cn] =
                (_Float16)(gelu * u);
        }
    }
}

// ---------------------------------------------------------------------------
// Kernel 2: down GEMM, 4-way K split, atomic f32 accumulation into out.
// grid = 64 column tiles x 4 K-splits = 256 blocks; 32-wide N tile.
// Same double-buffered single-barrier pipeline.
// ---------------------------------------------------------------------------
__global__ __launch_bounds__(256) void mlp_down_kernel(
    const _Float16* __restrict__ inter,
    const int*      __restrict__ dW, const float* __restrict__ dS,
    float*          __restrict__ out)
{
    __shared__ _Float16 ldsD[2][32 * LDSW];
    __shared__ unsigned lut256[256];

    const int tid = threadIdx.x;
    lut256[tid] = lut_entry(tid);

    const int n0    = (blockIdx.x & 63) * 32;
    const int kbase = (blockIdx.x >> 6) * (MLPD / 4);
    const int lane  = tid & 31;
    const int t0    = (tid >> 5) * 16;

    const int rbase = tid >> 4;            // dequant rows rbase + {0,16}
    const int j     = tid & 15;

    const int m     = lane & 15;
    const int khalf = lane >> 4;
    const int token = t0 + m;

    const int bn   = lane & 15;
    const int koff = (lane >> 4) * 16;

    size_t rw2[2], rs2[2];
#pragma unroll
    for (int s = 0; s < 2; ++s) {
        const int row = rbase + 16 * s;
        rw2[s] = (size_t)(n0 + row) * (MLPD / 2);
        rs2[s] = (size_t)(n0 + row) * (MLPD / 32);
    }

    v8f acc[2] = {{}, {}};

    // ---- prologue ----
    int   pd[2];
    float sd[2];
#pragma unroll
    for (int s = 0; s < 2; ++s) {
        pd[s] = dW[rw2[s] + (kbase >> 1) + j];
        sd[s] = dS[rs2[s] + (kbase >> 5)];
    }
    __syncthreads();
    {
        unsigned lw[2];
#pragma unroll
        for (int s = 0; s < 2; ++s) lw[s] = lut256[pd[s] & 255];
#pragma unroll
        for (int s = 0; s < 2; ++s) {
            const int row = rbase + 16 * s;
            *(v2h*)&ldsD[0][row * LDSW + 2 * j] = scale_mul(lw[s], sd[s]);
        }
    }
    __syncthreads();

    const int iters = (MLPD / 4) / 32;
    for (int it = 0; it < iters; ++it) {
        const int k0   = kbase + it * 32;
        const int cur  = it & 1;
        const bool more = (it + 1) < iters;

        if (more) {
#pragma unroll
            for (int s = 0; s < 2; ++s) {
                pd[s] = dW[rw2[s] + ((k0 + 32) >> 1) + j];
                sd[s] = dS[rs2[s] + ((k0 + 32) >> 5)];
            }
        }

        const _Float16* ip = inter + (size_t)token * MLPD + k0 + khalf * 16;
        const v8h a0 = *(const v8h*)ip;
        const v8h a1 = *(const v8h*)(ip + 8);
        v16h a;
#pragma unroll
        for (int i = 0; i < 8; ++i) { a[i] = a0[i]; a[8 + i] = a1[i]; }

#pragma unroll
        for (int nt = 0; nt < 2; ++nt) {
            const v16h b = *(const v16h*)&ldsD[cur][(nt * 16 + bn) * LDSW + koff];
            acc[nt] = __builtin_amdgcn_wmma_f32_16x16x32_f16(
                false, a, false, b, (short)0, acc[nt], false, false);
        }

        if (more) {
            unsigned lw[2];
#pragma unroll
            for (int s = 0; s < 2; ++s) lw[s] = lut256[pd[s] & 255];
#pragma unroll
            for (int s = 0; s < 2; ++s) {
                const int row = rbase + 16 * s;
                *(v2h*)&ldsD[cur ^ 1][row * LDSW + 2 * j] = scale_mul(lw[s], sd[s]);
            }
        }

        __syncthreads();
    }

    // ---- accumulate K-split partials into zero-initialized out ----
    const int cn    = lane & 15;
    const int mbase = t0 + ((lane >> 4) << 3);
#pragma unroll
    for (int nt = 0; nt < 2; ++nt)
#pragma unroll
        for (int rr = 0; rr < 8; ++rr)
            atomicAdd(&out[(size_t)(mbase + rr) * HID + n0 + nt * 16 + cn],
                      acc[nt][rr]);
}

// ---------------------------------------------------------------------------
extern "C" void kernel_launch(void* const* d_in, const int* in_sizes, int n_in,
                              void* d_out, int out_size, void* d_ws, size_t ws_size,
                              hipStream_t stream) {
    (void)in_sizes; (void)n_in; (void)ws_size;

    const float* x  = (const float*)d_in[0];
    const int*   gW = (const int*)  d_in[1];
    const float* gS = (const float*)d_in[2];
    const int*   uW = (const int*)  d_in[3];
    const float* uS = (const float*)d_in[4];
    const int*   dW = (const int*)  d_in[5];
    const float* dS = (const float*)d_in[6];

    _Float16* inter = (_Float16*)d_ws;                       // 4 MiB
    _Float16* xh    = (_Float16*)((char*)d_ws +
                                  (size_t)NTOK * MLPD * sizeof(_Float16));
    float*    out   = (float*)d_out;

    // out must start at zero: kernel 2 accumulates K-split partials.
    hipMemsetAsync(out, 0, (size_t)out_size * sizeof(float), stream);

    cvt_x_kernel     <<<NTOK * HID / (4 * 256), 256, 0, stream>>>(x, xh);
    mlp_gateup_kernel<<<MLPD / 64, 256, 0, stream>>>(xh, gW, gS, uW, uS, inter);
    mlp_down_kernel  <<<(HID / 32) * 4, 256, 0, stream>>>(inter, dW, dS, out);
}
